// AffineTransform_21285857919707
// MI455X (gfx1250) — matrix operand order repaired and checked by reading, not compiled, not run on hardware
//
#include <hip/hip_runtime.h>
#include <hip/hip_bf16.h>
#include <math.h>

typedef __attribute__((ext_vector_type(16))) _Float16 v16h;
typedef __attribute__((ext_vector_type(8)))  float    v8f;
typedef __attribute__((ext_vector_type(4)))  float    v4f;
typedef __attribute__((ext_vector_type(4)))  _Float16 v4h;

#define HEADS 6
#define HD    32
#define NTOK  512
#define KSTR  40      // padded f16 row stride for K / V^T tiles (bank stagger)
#define NPOS  43      // rel_idx range: 0..42

// ---------------------------------------------------------------------------
// DPP16 row_xmask butterfly (VALU-only cross-lane, no LDS port traffic).
// ---------------------------------------------------------------------------
template <int MASK>
__device__ __forceinline__ float dpp_xmask(float x) {
  int xi = __builtin_bit_cast(int, x);
  int r  = __builtin_amdgcn_update_dpp(xi, xi, 0x160 | MASK, 0xf, 0xf, false);
  return __builtin_bit_cast(float, r);
}
__device__ __forceinline__ float rowmax16(float x) {
  x = fmaxf(x, dpp_xmask<1>(x));
  x = fmaxf(x, dpp_xmask<2>(x));
  x = fmaxf(x, dpp_xmask<4>(x));
  x = fmaxf(x, dpp_xmask<8>(x));
  return x;
}

// ---------------------------------------------------------------------------
// Tiny kernel: dynamic position-bias MLP for the 43 reachable table rows.
// ---------------------------------------------------------------------------
__device__ __forceinline__ void ln_relu12(const float* in, float* out,
                                          const float* g, const float* b) {
  float mu = 0.f;
#pragma unroll
  for (int i = 0; i < 12; ++i) mu += in[i];
  mu *= (1.f / 12.f);
  float var = 0.f;
#pragma unroll
  for (int i = 0; i < 12; ++i) { float d = in[i] - mu; var += d * d; }
  var *= (1.f / 12.f);
  float rs = rsqrtf(var + 1e-5f);
#pragma unroll
  for (int i = 0; i < 12; ++i) {
    float t = (in[i] - mu) * rs * g[i] + b[i];
    out[i] = t > 0.f ? t : 0.f;
  }
}

__global__ __launch_bounds__(64) void pos_mlp_kernel(
    const float* __restrict__ pw,  const float* __restrict__ pb,
    const float* __restrict__ g1,  const float* __restrict__ bb1,
    const float* __restrict__ w1,  const float* __restrict__ b1,
    const float* __restrict__ g2,  const float* __restrict__ bb2,
    const float* __restrict__ w2,  const float* __restrict__ b2,
    const float* __restrict__ g3,  const float* __restrict__ bb3,
    const float* __restrict__ w3,  const float* __restrict__ b3,
    float* __restrict__ pos_out) {
  int r = threadIdx.x;
  if (r >= NPOS) return;
  int ci = r / 225, cj = (r / 15) % 15, ck = r % 15;
  float c0 = (float)(ci - 7), c1 = (float)(cj - 7), c2 = (float)(ck - 7);

  float x[12], y[12];
#pragma unroll
  for (int c = 0; c < 12; ++c)
    x[c] = c0 * pw[0 * 12 + c] + c1 * pw[1 * 12 + c] + c2 * pw[2 * 12 + c] + pb[c];

  ln_relu12(x, y, g1, bb1);
#pragma unroll
  for (int c = 0; c < 12; ++c) {
    float s = b1[c];
#pragma unroll
    for (int kk = 0; kk < 12; ++kk) s += y[kk] * w1[kk * 12 + c];
    x[c] = s;
  }
  ln_relu12(x, y, g2, bb2);
#pragma unroll
  for (int c = 0; c < 12; ++c) {
    float s = b2[c];
#pragma unroll
    for (int kk = 0; kk < 12; ++kk) s += y[kk] * w2[kk * 12 + c];
    x[c] = s;
  }
  ln_relu12(x, y, g3, bb3);
#pragma unroll
  for (int h = 0; h < HEADS; ++h) {
    float s = b3[h];
#pragma unroll
    for (int kk = 0; kk < 12; ++kk) s += y[kk] * w3[kk * HEADS + h];
    pos_out[r * HEADS + h] = s;
  }
}

// ---------------------------------------------------------------------------
// Fused flash attention. Grid: (B*H*4) blocks of 256 threads (8 wave32s).
// Each block: one (b,h) pair, a 128-row Q group; each wave a 16-row Q tile.
// kv loop over 32-key tiles with register-double-buffered staging: tile i+1's
// global loads are issued right after tile i becomes visible in LDS, hiding
// HBM/L2 latency behind the WMMA + softmax work of tile i.
// ---------------------------------------------------------------------------
__global__ __launch_bounds__(256) void attn_kernel(
    const float* __restrict__ q, const float* __restrict__ k,
    const float* __restrict__ v, const float* __restrict__ pos,
    float* __restrict__ out) {
  __shared__ _Float16 sK [32 * KSTR];   // [key_row][c]
  __shared__ _Float16 sVT[32 * KSTR];   // [head_col][key_row]
  __shared__ _Float16 sP [8][16 * 32];  // per-wave P tile (C->A layout bounce)
  __shared__ float    sPos[NPOS * 8];

  const int lane = threadIdx.x & 31;
  const int wv   = threadIdx.x >> 5;
  const int l15  = lane & 15;
  const int hi   = (lane >> 4) & 1;     // which 16-lane half

  const int bh    = blockIdx.x >> 2;
  const int tg    = blockIdx.x & 3;
  const int head  = bh % HEADS;
  const int qbase = tg * 128 + wv * 16;
  const size_t base = (size_t)bh * NTOK * HD;

  for (int i = threadIdx.x; i < NPOS * HEADS; i += 256)
    sPos[(i / HEADS) * 8 + (i % HEADS)] = pos[i];

  const float scale = 0.17677669529663687f;  // 32^-0.5

  // ---- Q A-fragment (16x32 f16, scale folded in) -------------------------
  v16h aq;
  {
    const float* qp = q + base + (size_t)(qbase + l15) * HD;
    const int c0 = hi ? 16 : 0;
#pragma unroll
    for (int i = 0; i < 8; ++i) {
      aq[i]     = (_Float16)(qp[c0 + i] * scale);
      aq[8 + i] = (_Float16)(qp[c0 + 8 + i] * scale);
    }
  }

  // all-ones B fragment for the row-sum WMMA
  v16h bones;
#pragma unroll
  for (int i = 0; i < 16; ++i) bones[i] = (_Float16)1.0f;

  // coords of this lane's 8 C-layout rows
  int hn[8], wn[8], dn[8];
#pragma unroll
  for (int j = 0; j < 8; ++j) {
    int n = qbase + j + hi * 8;
    hn[j] = n >> 6; wn[j] = (n >> 3) & 7; dn[j] = n & 7;
  }

  v8f o0 = {}; v8f o1 = {}; v8f lacc = {};
  float mrow[8];
#pragma unroll
  for (int j = 0; j < 8; ++j) mrow[j] = -INFINITY;

  // staging geometry: one float4 chunk per thread (256 chunks = 32x32 tile)
  const int sr  = threadIdx.x >> 3;
  const int sc4 = (threadIdx.x & 7) * 4;

  // ---- pipeline prologue: load tile kv=0 into registers -------------------
  v4f kf = *(const v4f*)(k + base + sr * HD + sc4);
  v4f vf = *(const v4f*)(v + base + sr * HD + sc4);

  for (int kv = 0; kv < NTOK; kv += 32) {
    __syncthreads();  // prior-iteration LDS reads complete before restage

    // ---- convert + store the register-buffered tile into LDS -------------
    {
      v4h kh;
      kh[0] = (_Float16)kf[0]; kh[1] = (_Float16)kf[1];
      kh[2] = (_Float16)kf[2]; kh[3] = (_Float16)kf[3];
      *(v4h*)&sK[sr * KSTR + sc4] = kh;
      sVT[(sc4 + 0) * KSTR + sr] = (_Float16)vf[0];
      sVT[(sc4 + 1) * KSTR + sr] = (_Float16)vf[1];
      sVT[(sc4 + 2) * KSTR + sr] = (_Float16)vf[2];
      sVT[(sc4 + 3) * KSTR + sr] = (_Float16)vf[3];
    }
    __syncthreads();

    // ---- issue next tile's global loads (consumed next iteration) --------
    if (kv + 32 < NTOK) {
      const float* kp = k + base + (size_t)(kv + 32) * HD;
      const float* vp = v + base + (size_t)(kv + 32) * HD;
      kf = *(const v4f*)(kp + sr * HD + sc4);
      vf = *(const v4f*)(vp + sr * HD + sc4);
      __builtin_prefetch(kp + 32 * HD, 0, 0);   // warm tile i+2
      __builtin_prefetch(vp + 32 * HD, 0, 0);
    }

    // ---- S = Q * K^T for two 16-key chunks -------------------------------
    v16h bk0, bk1;
    {
      const int cc = hi ? 16 : 0;
#pragma unroll
      for (int i = 0; i < 16; ++i) {
        bk0[i] = sK[(l15)      * KSTR + cc + i];
        bk1[i] = sK[(16 + l15) * KSTR + cc + i];
      }
    }
    v8f zero = {};
    v8f s0 = __builtin_amdgcn_wmma_f32_16x16x32_f16(false, aq, false, bk0,
                                                    (short)0, zero, false, false);
    v8f s1 = __builtin_amdgcn_wmma_f32_16x16x32_f16(false, aq, false, bk1,
                                                    (short)0, zero, false, false);

    // ---- position bias + online softmax (max via DPP row_xmask) ----------
    const int m0 = kv + l15, m1 = m0 + 16;
    const int hm0 = m0 >> 6, wm0 = (m0 >> 3) & 7, dm0 = m0 & 7;
    const int hm1 = m1 >> 6, wm1 = (m1 >> 3) & 7, dm1 = m1 & 7;
    float p0v[8], p1v[8], alpha[8];
#pragma unroll
    for (int j = 0; j < 8; ++j) {
      int i0 = (hn[j] - hm0) + (wn[j] - wm0) + (dn[j] - dm0) + 21;
      int i1 = (hn[j] - hm1) + (wn[j] - wm1) + (dn[j] - dm1) + 21;
      float t0 = s0[j] + sPos[i0 * 8 + head];
      float t1 = s1[j] + sPos[i1 * 8 + head];
      float mx    = rowmax16(fmaxf(t0, t1));
      float mnew  = fmaxf(mrow[j], mx);
      alpha[j]    = __expf(mrow[j] - mnew);
      mrow[j]     = mnew;
      p0v[j] = __expf(t0 - mnew);
      p1v[j] = __expf(t1 - mnew);
      o0[j] *= alpha[j]; o1[j] *= alpha[j]; lacc[j] *= alpha[j];
    }

    // ---- C-layout -> A-layout bounce of P through per-wave LDS -----------
    // LDS ops from the same wave complete in order (ISA 7.3); only a compiler
    // reordering fence is needed, no workgroup barrier.
    {
      _Float16* pwl = &sP[wv][0];
#pragma unroll
      for (int j = 0; j < 8; ++j) {
        int row = j + hi * 8;
        pwl[row * 32 + l15]      = (_Float16)p0v[j];
        pwl[row * 32 + 16 + l15] = (_Float16)p1v[j];
      }
    }
    __asm volatile("" ::: "memory");

    v16h ap, bv0, bv1;
    {
      const _Float16* pr = &sP[wv][0];
      const int k0 = hi ? 16 : 0;
#pragma unroll
      for (int i = 0; i < 8; ++i) {
        ap[i]     = pr[l15 * 32 + k0 + i];
        ap[8 + i] = pr[l15 * 32 + k0 + 8 + i];
      }
      const int cc = hi ? 16 : 0;
#pragma unroll
      for (int i = 0; i < 16; ++i) {
        bv0[i] = sVT[(l15)      * KSTR + cc + i];
        bv1[i] = sVT[(16 + l15) * KSTR + cc + i];
      }
    }
    // O += P*V (two head-dim chunks);  l += P*ones (row-sum in C layout)
    o0   = __builtin_amdgcn_wmma_f32_16x16x32_f16(false, ap, false, bv0,
                                                  (short)0, o0, false, false);
    o1   = __builtin_amdgcn_wmma_f32_16x16x32_f16(false, ap, false, bv1,
                                                  (short)0, o1, false, false);
    lacc = __builtin_amdgcn_wmma_f32_16x16x32_f16(false, ap, false, bones,
                                                  (short)0, lacc, false, false);
  }

  // ---- normalize and store ------------------------------------------------
  float* op = out + base + (size_t)qbase * HD;
#pragma unroll
  for (int j = 0; j < 8; ++j) {
    int row = j + hi * 8;
    float inv = 1.f / lacc[j];
    op[row * HD + l15]      = o0[j] * inv;
    op[row * HD + 16 + l15] = o1[j] * inv;
  }
}

// ---------------------------------------------------------------------------
extern "C" void kernel_launch(void* const* d_in, const int* in_sizes, int n_in,
                              void* d_out, int out_size, void* d_ws, size_t ws_size,
                              hipStream_t stream) {
  (void)in_sizes; (void)n_in; (void)out_size; (void)ws_size;
  const float* q   = (const float*)d_in[0];
  const float* k   = (const float*)d_in[1];
  const float* v   = (const float*)d_in[2];
  // d_in[3..5] = h, w, d scalars (hardcoded WS=8)
  const float* pw  = (const float*)d_in[6];
  const float* pb  = (const float*)d_in[7];
  const float* g1  = (const float*)d_in[8];
  const float* bb1 = (const float*)d_in[9];
  const float* w1  = (const float*)d_in[10];
  const float* b1  = (const float*)d_in[11];
  const float* g2  = (const float*)d_in[12];
  const float* bb2 = (const float*)d_in[13];
  const float* w2  = (const float*)d_in[14];
  const float* b2  = (const float*)d_in[15];
  const float* g3  = (const float*)d_in[16];
  const float* bb3 = (const float*)d_in[17];
  const float* w3  = (const float*)d_in[18];
  const float* b3  = (const float*)d_in[19];
  float* pos = (float*)d_ws;  // NPOS*HEADS floats

  pos_mlp_kernel<<<1, 64, 0, stream>>>(pw, pb, g1, bb1, w1, b1, g2, bb2,
                                       w2, b2, g3, bb3, w3, b3, pos);
  attn_kernel<<<64 * HEADS * 4, 256, 0, stream>>>(q, k, v, pos, (float*)d_out);
}
